// MHAttention_20822001451495
// MI455X (gfx1250) — matrix operand order repaired and checked by reading, not compiled, not run on hardware
//
#include <hip/hip_runtime.h>
#include <hip/hip_bf16.h>

typedef __attribute__((ext_vector_type(16))) __bf16 v16bf;
typedef __attribute__((ext_vector_type(8)))  __bf16 bf8v;
typedef __attribute__((ext_vector_type(8)))  float  v8f;
typedef int v4i_g __attribute__((vector_size(16)));   // matches builtin param type

#define BQ 4
#define TQ 2048
#define CQ 2048
#define HQ 16
#define DHQ 128
#define BH (BQ*HQ)      // 64
#define MROWS (BQ*TQ)   // 8192
#define N3C (3*CQ)      // 6144

#if defined(__has_builtin)
#if __has_builtin(__builtin_amdgcn_global_load_async_to_lds_b128)
#define HAVE_ASYNC_B128 1
#endif
#if __has_builtin(__builtin_amdgcn_s_wait_asynccnt)
#define HAVE_WAIT_ASYNC 1
#endif
#endif

__device__ __forceinline__ v8f wmma_bf16(v16bf a, v16bf b, v8f c) {
  return __builtin_amdgcn_wmma_f32_16x16x32_bf16(false, a, false, b, (short)0, c, false, false);
}

// A-fragment (16x32 bf16) from row-major [., ld] matrix.
// lanes 0-15: row = row0+lane, K {k0..k0+7, k0+16..k0+23}
// lanes 16-31: row = row0+lane-16, K {k0+8..k0+15, k0+24..k0+31}
__device__ __forceinline__ v16bf load_a_frag(const __bf16* base, int ld, int row0, int k0, int lane) {
  int half = lane >> 4, l = lane & 15;
  const __bf16* p = base + (long)(row0 + l) * ld + k0 + half * 8;
  union { v16bf v; bf8v h[2]; } u;
  u.h[0] = *(const bf8v*)(p);
  u.h[1] = *(const bf8v*)(p + 16);
  return u.v;
}

// B-fragment (32x16 bf16, KxN) where source is row-major [N, K] (i.e. column n
// of B is contiguous in memory): lanes 0-15: n = n0+lane, K {k0..k0+7, k0+16..k0+23};
// lanes 16-31: n = n0+lane-16, K {k0+8..k0+15, k0+24..k0+31}.
__device__ __forceinline__ v16bf load_b_frag(const __bf16* base, int ld, int n0, int k0, int lane) {
  int half = lane >> 4, l = lane & 15;
  const __bf16* p = base + (long)(n0 + l) * ld + k0 + half * 8;
  union { v16bf v; bf8v h[2]; } u;
  u.h[0] = *(const bf8v*)(p);
  u.h[1] = *(const bf8v*)(p + 16);
  return u.v;
}

// --------- async global->LDS (ASYNCcnt path; sync fallback) ----------
__device__ __forceinline__ void async_copy_b128(__bf16* lds_dst, const __bf16* gsrc) {
#if defined(HAVE_ASYNC_B128)
  __builtin_amdgcn_global_load_async_to_lds_b128(
      (__attribute__((address_space(1))) v4i_g*)gsrc,
      (__attribute__((address_space(3))) v4i_g*)lds_dst, 0, 0);
#else
  *(uint4*)lds_dst = *(const uint4*)gsrc;
#endif
}

__device__ __forceinline__ void wait_async_all() {
#if defined(HAVE_WAIT_ASYNC)
  __builtin_amdgcn_s_wait_asynccnt(0);
#elif defined(HAVE_ASYNC_B128)
  asm volatile("s_wait_asynccnt 0" ::: "memory");
#endif
}

__device__ __forceinline__ void wait_async_keep2() {
#if defined(HAVE_WAIT_ASYNC)
  __builtin_amdgcn_s_wait_asynccnt(2);
#elif defined(HAVE_ASYNC_B128)
  asm volatile("s_wait_asynccnt 2" ::: "memory");
#endif
}

// ---------------- fp32 -> bf16 conversion ----------------
__global__ __launch_bounds__(256) void cvt_bf16(const float* __restrict__ src,
                                                __bf16* __restrict__ dst, long n) {
  long i = ((long)blockIdx.x * 256 + threadIdx.x) * 4;
  if (i + 3 < n) {
    float4 f = *(const float4*)(src + i);
    dst[i + 0] = (__bf16)f.x; dst[i + 1] = (__bf16)f.y;
    dst[i + 2] = (__bf16)f.z; dst[i + 3] = (__bf16)f.w;
  } else {
    for (long j = i; j < n; ++j) dst[j] = (__bf16)src[j];
  }
}

// ---------------- QKV GEMM: qkv = x @ qkv_w^T + b ----------------
__global__ __launch_bounds__(32) void qkv_gemm(const __bf16* __restrict__ x,
                                               const __bf16* __restrict__ w,
                                               const float* __restrict__ bias,
                                               __bf16* __restrict__ qb,
                                               __bf16* __restrict__ kb,
                                               __bf16* __restrict__ vtb) {
  const int lane = threadIdx.x;
  const int MT = MROWS / 64;                 // 128
  int mt = blockIdx.x % MT, nt = blockIdx.x / MT;
  int m0 = mt * 64, n0 = nt * 64;
  v8f acc[4][4] = {};
  for (int k0 = 0; k0 < CQ; k0 += 32) {
    v16bf a[4], b[4];
#pragma unroll
    for (int i = 0; i < 4; ++i) a[i] = load_a_frag(x, CQ, m0 + i * 16, k0, lane);
#pragma unroll
    for (int j = 0; j < 4; ++j) b[j] = load_b_frag(w, CQ, n0 + j * 16, k0, lane);
#pragma unroll
    for (int i = 0; i < 4; ++i)
#pragma unroll
      for (int j = 0; j < 4; ++j) acc[i][j] = wmma_bf16(a[i], b[j], acc[i][j]);
  }
  int half = lane >> 4, l = lane & 15;
#pragma unroll
  for (int i = 0; i < 4; ++i)
#pragma unroll
    for (int j = 0; j < 4; ++j) {
      int col = n0 + j * 16 + l;
      int which = col >> 11;          // 0:q 1:k 2:v
      int c = col & (CQ - 1);
      int h = c >> 7, d = c & 127;
#pragma unroll
      for (int r = 0; r < 8; ++r) {
        int row = m0 + i * 16 + r + half * 8;
        int bb = row >> 11, t = row & (TQ - 1);
        int bh = bb * HQ + h;
        __bf16 bv = (__bf16)(acc[i][j][r] + bias[col]);
        if (which == 0)      qb[((long)bh * TQ + t) * DHQ + d] = bv;
        else if (which == 1) kb[((long)bh * TQ + t) * DHQ + d] = bv;
        else                 vtb[((long)bh * DHQ + d) * TQ + t] = bv;
      }
    }
}

// ---------------- Flash attention (causal, online softmax) ----------------
// 8 waves / block, each wave owns 16 query rows; key tiles of 32 are staged
// into LDS once per block via async-to-LDS (double buffered), shared by all
// waves. Key-loop trip count is uniform per block; out-of-causal-range tiles
// are fully masked (exp -> 0), so sharing is exact.
__global__ __launch_bounds__(256) void attn_fwd(const __bf16* __restrict__ Q,
                                                const __bf16* __restrict__ K,
                                                const __bf16* __restrict__ VT,
                                                __bf16* __restrict__ out) {
  __shared__ __bf16 kt[2][32][DHQ];          // 2 x 8KB  : keys x dh
  __shared__ __bf16 vtile[2][DHQ][32];       // 2 x 8KB  : dh x keys
  __shared__ __bf16 plds[8][16][40];         // per-wave P transpose scratch
  const int tid = threadIdx.x;
  const int lane = tid & 31;
  const int wave = tid >> 5;
  const int qb = blockIdx.x & 15;            // T/128 = 16 query blocks
  const int bh = blockIdx.x >> 4;            // 0..63
  const __bf16* Qp = Q + (long)bh * TQ * DHQ;
  const __bf16* Kp = K + (long)bh * TQ * DHQ;
  const __bf16* Vp = VT + (long)bh * DHQ * TQ;
  const int q0 = qb * 128 + wave * 16;
  const int half = lane >> 4, l = lane & 15;
  const int ksteps = (qb + 1) * 4;           // keys 0 .. qb*128+127

  v16bf qf[4];
#pragma unroll
  for (int c = 0; c < 4; ++c) qf[c] = load_a_frag(Qp, DHQ, q0, c * 32, lane);

  v8f o[8] = {};
  float m[8], lsum[8], alpha[8];
#pragma unroll
  for (int r = 0; r < 8; ++r) { m[r] = -3.0e38f; lsum[r] = 0.f; }

  const float scale = 0.08838834764831845f;  // 1/sqrt(128)
  const float L2E = 1.4426950408889634f;

  // per-thread slices of the 8KB tiles (256 threads x 32B each)
  const int vrow = tid >> 1, vhalf = tid & 1;
  auto issue_tile = [&](int ks, int buf) {
    async_copy_b128(&kt[buf][0][0] + tid * 16, Kp + (long)ks * 32 * DHQ + tid * 16);
    async_copy_b128(&vtile[buf][vrow][vhalf * 16],
                    Vp + (long)vrow * TQ + ks * 32 + vhalf * 16);
  };

  issue_tile(0, 0);
  for (int ks = 0; ks < ksteps; ++ks) {
    const int buf = ks & 1;
    if (ks + 1 < ksteps) {
      issue_tile(ks + 1, buf ^ 1);
      wait_async_keep2();                    // current tile landed; prefetch in flight
    } else {
      wait_async_all();
    }
    __syncthreads();

    const int kb0 = ks * 32;
    v8f s0 = {}, s1 = {};
#pragma unroll
    for (int c = 0; c < 4; ++c) {
      v16bf b0 = load_b_frag(&kt[buf][0][0], DHQ, 0, c * 32, lane);
      v16bf b1 = load_b_frag(&kt[buf][0][0], DHQ, 16, c * 32, lane);
      s0 = wmma_bf16(qf[c], b0, s0);
      s1 = wmma_bf16(qf[c], b1, s1);
    }
    float p0[8], p1[8];
#pragma unroll
    for (int r = 0; r < 8; ++r) {
      int qrow = q0 + r + half * 8;
      int key0 = kb0 + l, key1 = kb0 + 16 + l;
      float v0 = (key0 <= qrow) ? s0[r] * scale : -3.0e38f;
      float v1 = (key1 <= qrow) ? s1[r] * scale : -3.0e38f;
      float mx = fmaxf(v0, v1);
#pragma unroll
      for (int d = 1; d < 16; d <<= 1) mx = fmaxf(mx, __shfl_xor(mx, d, 32));
      float mn = fmaxf(m[r], mx);
      float a_ = __builtin_amdgcn_exp2f((m[r] - mn) * L2E);
      p0[r] = __builtin_amdgcn_exp2f((v0 - mn) * L2E);
      p1[r] = __builtin_amdgcn_exp2f((v1 - mn) * L2E);
      float rs = p0[r] + p1[r];
#pragma unroll
      for (int d = 1; d < 16; d <<= 1) rs += __shfl_xor(rs, d, 32);
      lsum[r] = lsum[r] * a_ + rs;
      m[r] = mn;
      alpha[r] = a_;
    }
    // C-layout -> A-layout transpose of P through per-wave LDS scratch
#pragma unroll
    for (int r = 0; r < 8; ++r) {
      plds[wave][r + half * 8][l] = (__bf16)p0[r];
      plds[wave][r + half * 8][16 + l] = (__bf16)p1[r];
    }
    asm volatile("s_wait_dscnt 0" ::: "memory");
    v16bf pf = load_a_frag(&plds[wave][0][0], 40, 0, 0, lane);
#pragma unroll
    for (int n = 0; n < 8; ++n) {
#pragma unroll
      for (int r = 0; r < 8; ++r) o[n][r] *= alpha[r];
      v16bf vfrag = load_b_frag(&vtile[buf][0][0], 32, n * 16, 0, lane);
      o[n] = wmma_bf16(pf, vfrag, o[n]);
    }
    __syncthreads();                          // all reads of buf done before reuse
  }
  const int bb = bh >> 4, h = bh & 15;
#pragma unroll
  for (int n = 0; n < 8; ++n)
#pragma unroll
    for (int r = 0; r < 8; ++r) {
      int qrow = q0 + r + half * 8;
      float v = o[n][r] / lsum[r];
      out[((long)(bb * TQ + qrow)) * CQ + h * DHQ + n * 16 + l] = (__bf16)v;
    }
}

// ---------------- proj GEMM: out = attn @ proj_w^T + b (fp32 out) ----------------
__global__ __launch_bounds__(32) void proj_gemm(const __bf16* __restrict__ a,
                                                const __bf16* __restrict__ w,
                                                const float* __restrict__ bias,
                                                float* __restrict__ out) {
  const int lane = threadIdx.x;
  const int MT = MROWS / 64;                 // 128
  int mt = blockIdx.x % MT, nt = blockIdx.x / MT;
  int m0 = mt * 64, n0 = nt * 64;
  v8f acc[4][4] = {};
  for (int k0 = 0; k0 < CQ; k0 += 32) {
    v16bf af[4], bf[4];
#pragma unroll
    for (int i = 0; i < 4; ++i) af[i] = load_a_frag(a, CQ, m0 + i * 16, k0, lane);
#pragma unroll
    for (int j = 0; j < 4; ++j) bf[j] = load_b_frag(w, CQ, n0 + j * 16, k0, lane);
#pragma unroll
    for (int i = 0; i < 4; ++i)
#pragma unroll
      for (int j = 0; j < 4; ++j) acc[i][j] = wmma_bf16(af[i], bf[j], acc[i][j]);
  }
  int half = lane >> 4, l = lane & 15;
#pragma unroll
  for (int i = 0; i < 4; ++i)
#pragma unroll
    for (int j = 0; j < 4; ++j) {
      int col = n0 + j * 16 + l;
      float b = bias[col];
#pragma unroll
      for (int r = 0; r < 8; ++r) {
        int row = m0 + i * 16 + r + half * 8;
        out[(long)row * CQ + col] = acc[i][j][r] + b;
      }
    }
}

extern "C" void kernel_launch(void* const* d_in, const int* in_sizes, int n_in,
                              void* d_out, int out_size, void* d_ws, size_t ws_size,
                              hipStream_t stream) {
  const float* x      = (const float*)d_in[0];
  const float* qkv_w  = (const float*)d_in[1];
  const float* qkv_b  = (const float*)d_in[2];
  const float* proj_w = (const float*)d_in[3];
  const float* proj_b = (const float*)d_in[4];
  float* out = (float*)d_out;

  char* ws = (char*)d_ws;
  const long N_X   = (long)MROWS * CQ;       // 16,777,216
  const long N_QW  = (long)N3C * CQ;         // 12,582,912
  const long N_PW  = (long)CQ * CQ;          // 4,194,304
  const long N_HEA = (long)BH * TQ * DHQ;    // 16,777,216

  __bf16* x_bf   = (__bf16*)ws;                 ws += N_X * 2;
  __bf16* qw_bf  = (__bf16*)ws;                 ws += N_QW * 2;
  __bf16* pw_bf  = (__bf16*)ws;                 ws += N_PW * 2;
  __bf16* q_bf   = (__bf16*)ws;                 ws += N_HEA * 2;
  __bf16* k_bf   = (__bf16*)ws;                 ws += N_HEA * 2;
  __bf16* vt_bf  = (__bf16*)ws;                 ws += N_HEA * 2;
  __bf16* att_bf = (__bf16*)ws;                 ws += N_X * 2;

  auto cvt_blocks = [](long n) { return (unsigned)((n + 1023) / 1024); };
  cvt_bf16<<<cvt_blocks(N_X),  256, 0, stream>>>(x,      x_bf,  N_X);
  cvt_bf16<<<cvt_blocks(N_QW), 256, 0, stream>>>(qkv_w,  qw_bf, N_QW);
  cvt_bf16<<<cvt_blocks(N_PW), 256, 0, stream>>>(proj_w, pw_bf, N_PW);

  qkv_gemm<<<dim3((MROWS / 64) * (N3C / 64)), dim3(32), 0, stream>>>(
      x_bf, qw_bf, qkv_b, q_bf, k_bf, vt_bf);

  attn_fwd<<<dim3(BH * (TQ / 128)), dim3(256), 0, stream>>>(q_bf, k_bf, vt_bf, att_bf);

  proj_gemm<<<dim3((MROWS / 64) * (CQ / 64)), dim3(32), 0, stream>>>(
      att_bf, pw_bf, proj_b, out);

  (void)in_sizes; (void)n_in; (void)out_size; (void)ws_size;
}
